// VisionMamba_21088289424015
// MI455X (gfx1250) — matrix operand order repaired and checked by reading, not compiled, not run on hardware
//
#include <hip/hip_runtime.h>
#include <hip/hip_bf16.h>

// ---------------- model dims ----------------
#define BDIM   4
#define IMG    224
#define PATCH  16
#define CCH    3
#define GRID_  14            // IMG/PATCH
#define LTOK   196           // GRID_*GRID_
#define DMODEL 768
#define DEPTH_ 12
#define NSTATE 16
#define KCONV  4
#define RRANK  48
#define RPAD   64            // dtr padded K (multiple of 32)
#define XDW    (RRANK + 2*NSTATE)   // 80
#define HMLP   2048
#define MROWS  (BDIM*LTOK)   // 784  (multiple of 16)

typedef __attribute__((ext_vector_type(16))) _Float16 v16h;
typedef __attribute__((ext_vector_type(8)))  _Float16 v8h;
typedef __attribute__((ext_vector_type(4)))  _Float16 v4h;
typedef __attribute__((ext_vector_type(8)))  float    v8f;
typedef __attribute__((ext_vector_type(4)))  float    v4f;

// load 16 halfs as two b128 loads, concatenated into one fragment register set
__device__ __forceinline__ v16h ldfrag(const _Float16* p0, const _Float16* p1)
{
    v8h lo = *(const v8h*)p0;
    v8h hi = *(const v8h*)p1;
    return __builtin_shufflevector(lo, hi, 0,1,2,3,4,5,6,7,8,9,10,11,12,13,14,15);
}

#define WMMA(a, b, c) __builtin_amdgcn_wmma_f32_16x16x32_f16(false, a, false, b, (short)0, c, false, false)

// =====================================================================
// WMMA GEMM (f16 operands, f32 accumulate):
//   Out[m,n] = sum_k A[m,k]*W[n,k]  (+bias[n]) (+residual)
// REQUIRES Kd % 32 == 0 and lda/ldw multiples of 8 (16B-aligned rows).
// One wave computes a 32x64 strip: 2 A-frags x 4 B-frags = 8 independent
// v_wmma_f32_16x16x32_f16 per K-step (covers the WMMA->WMMA hazard window),
// with software-pipelined (double-buffered) fragment loads so global latency
// overlaps the matrix pipe.  All tail predicates are wave-uniform (EXEC all-1).
// A/B fragments use contiguous k = 16*hi + s per lane half (wave32).
// =====================================================================
__global__ __launch_bounds__(256) void gemm_wmma_f16(
    const _Float16* __restrict__ A, int lda,
    const _Float16* __restrict__ W, int ldw,
    const float* __restrict__ bias,
    const float* __restrict__ residual,
    float* __restrict__ Out, int ldo,
    int M, int Nd, int Kd)
{
    const int lane = threadIdx.x & 31;
    const int hi   = lane >> 4;
    const int lo   = lane & 15;
    const int tilesM2 = (M + 31) >> 5;           // 32-row strips
    const int tilesN4 = (Nd + 63) >> 6;          // 64-col strips
    const int nTiles  = tilesM2 * tilesN4;
    int       tile  = blockIdx.x * (blockDim.x >> 5) + (threadIdx.x >> 5);
    const int tstep = gridDim.x * (blockDim.x >> 5);

    for (; tile < nTiles; tile += tstep) {
        const int tm2 = tile / tilesN4;
        const int tn4 = tile - tm2 * tilesN4;
        const int m0 = tm2 * 32, n0 = tn4 * 64;
        const bool exm = (m0 + 16) < M;                       // wave-uniform
        const _Float16* ar0 = A + (size_t)(m0 + lo) * lda;
        const _Float16* ar1 = A + (size_t)((exm ? m0 + 16 : m0) + lo) * lda;
        const _Float16* wr[4];
        bool exn[4];
#pragma unroll
        for (int j = 0; j < 4; ++j) {
            exn[j] = (n0 + j * 16) < Nd;                      // wave-uniform
            wr[j]  = W + (size_t)((exn[j] ? n0 + j * 16 : n0) + lo) * ldw;
        }

        v8f acc[8] = {};   // [mi*4 + j]

        // ---- prologue: fragments for K-step 0 ----
        v16h a0 = ldfrag(ar0 + hi * 16,       ar0 + hi * 16 + 8);
        v16h a1 = ldfrag(ar1 + hi * 16,       ar1 + hi * 16 + 8);
        v16h b0 = ldfrag(wr[0] + hi * 16,     wr[0] + hi * 16 + 8);
        v16h b1 = ldfrag(wr[1] + hi * 16,     wr[1] + hi * 16 + 8);
        v16h b2 = ldfrag(wr[2] + hi * 16,     wr[2] + hi * 16 + 8);
        v16h b3 = ldfrag(wr[3] + hi * 16,     wr[3] + hi * 16 + 8);

        for (int k0 = 0; k0 < Kd; k0 += 32) {
            const int kn = (k0 + 32 < Kd) ? (k0 + 32) : k0;   // branchless last iter
            __builtin_prefetch(ar0 + kn + 256, 0, 1);         // global_prefetch_b8
            __builtin_prefetch(wr[0] + kn + 256, 0, 1);

            // ---- load next K-step while WMMAs run on current ----
            v16h a0n = ldfrag(ar0 + kn + hi * 16,   ar0 + kn + hi * 16 + 8);
            v16h a1n = ldfrag(ar1 + kn + hi * 16,   ar1 + kn + hi * 16 + 8);
            v16h b0n = ldfrag(wr[0] + kn + hi * 16, wr[0] + kn + hi * 16 + 8);
            v16h b1n = ldfrag(wr[1] + kn + hi * 16, wr[1] + kn + hi * 16 + 8);
            v16h b2n = ldfrag(wr[2] + kn + hi * 16, wr[2] + kn + hi * 16 + 8);
            v16h b3n = ldfrag(wr[3] + kn + hi * 16, wr[3] + kn + hi * 16 + 8);

            acc[0] = WMMA(a0, b0, acc[0]);
            acc[1] = WMMA(a0, b1, acc[1]);
            acc[2] = WMMA(a0, b2, acc[2]);
            acc[3] = WMMA(a0, b3, acc[3]);
            acc[4] = WMMA(a1, b0, acc[4]);
            acc[5] = WMMA(a1, b1, acc[5]);
            acc[6] = WMMA(a1, b2, acc[6]);
            acc[7] = WMMA(a1, b3, acc[7]);

            a0 = a0n; a1 = a1n; b0 = b0n; b1 = b1n; b2 = b2n; b3 = b3n;
        }

        // ---- store: VGPR r holds M = hi*8+r, N = lo (16x16 C/D layout) ----
#pragma unroll
        for (int mi = 0; mi < 2; ++mi) {
            if (mi && !exm) continue;                         // wave-uniform
#pragma unroll
            for (int j = 0; j < 4; ++j) {
                if (!exn[j]) continue;                        // wave-uniform
                const int n  = n0 + j * 16 + lo;
                const float bv = bias ? bias[n] : 0.0f;
                const v8f  a  = acc[mi * 4 + j];
#pragma unroll
                for (int r = 0; r < 8; ++r) {
                    const int m = m0 + mi * 16 + hi * 8 + r;
                    float v = a[r] + bv;
                    if (residual) v += residual[(size_t)m * ldo + n];
                    Out[(size_t)m * ldo + n] = v;
                }
            }
        }
    }
}

// ---------------- f32 -> f16 vectorized convert (count % 4 == 0) ----------------
__global__ void cvt_f16_k(const float* __restrict__ src, _Float16* __restrict__ dst, int count4)
{
    int idx = blockIdx.x * blockDim.x + threadIdx.x;
    if (idx >= count4) return;
    v4f v = ((const v4f*)src)[idx];
    v4h h;
#pragma unroll
    for (int i = 0; i < 4; ++i) h[i] = (_Float16)v[i];
    ((v4h*)dst)[idx] = h;
}

// ---------------- zero-padded f32 -> f16 copy (pads K=48 up to 64) ----------------
__global__ void pad_f16_k(const float* __restrict__ src, int lds, int cols,
                          _Float16* __restrict__ dst, int ldd, int count)
{
    int idx = blockIdx.x * blockDim.x + threadIdx.x;
    if (idx >= count) return;
    int r = idx / ldd, c = idx - r * ldd;
    dst[idx] = (c < cols) ? (_Float16)src[(size_t)r * lds + c] : (_Float16)0.0f;
}

// =====================================================================
// LayerNorm over DMODEL, one 256-thread block per row.
// Writes f32 (outf) and/or f16 (outh) — whichever is non-null.
// =====================================================================
__global__ __launch_bounds__(256) void layernorm_k(
    const float* __restrict__ x, const float* __restrict__ w,
    const float* __restrict__ b, float* __restrict__ outf,
    _Float16* __restrict__ outh)
{
    __shared__ float red[256];
    const int row = blockIdx.x;
    const float* xr = x + (size_t)row * DMODEL;

    float s = 0.f;
    for (int i = threadIdx.x; i < DMODEL; i += 256) s += xr[i];
    red[threadIdx.x] = s; __syncthreads();
    for (int off = 128; off > 0; off >>= 1) {
        if (threadIdx.x < off) red[threadIdx.x] += red[threadIdx.x + off];
        __syncthreads();
    }
    const float mu = red[0] * (1.0f / DMODEL);
    __syncthreads();

    float v = 0.f;
    for (int i = threadIdx.x; i < DMODEL; i += 256) { float t = xr[i] - mu; v += t * t; }
    red[threadIdx.x] = v; __syncthreads();
    for (int off = 128; off > 0; off >>= 1) {
        if (threadIdx.x < off) red[threadIdx.x] += red[threadIdx.x + off];
        __syncthreads();
    }
    const float rstd = rsqrtf(red[0] * (1.0f / DMODEL) + 1e-5f);

    for (int i = threadIdx.x; i < DMODEL; i += 256) {
        float o = (xr[i] - mu) * rstd * w[i] + b[i];
        if (outf) outf[(size_t)row * DMODEL + i] = o;
        if (outh) outh[(size_t)row * DMODEL + i] = (_Float16)o;
    }
}

// ---------------- patch gather -> f16 GEMM operand ----------------
__global__ void patch_gather_k(const float* __restrict__ x, _Float16* __restrict__ xr)
{
    int idx = blockIdx.x * blockDim.x + threadIdx.x;
    if (idx >= MROWS * DMODEL) return;
    int col = idx % DMODEL;
    int m   = idx / DMODEL;
    int b = m / LTOK, l = m % LTOK;
    int gy = l / GRID_, gx = l % GRID_;
    int c = col / (PATCH * PATCH);
    int r = col % (PATCH * PATCH);
    int py = r / PATCH, px = r % PATCH;
    xr[idx] = (_Float16)x[(((size_t)(b * CCH + c)) * IMG + (gy * PATCH + py)) * IMG + (gx * PATCH + px)];
}

__global__ void add_pos_k(float* __restrict__ hs, const float* __restrict__ pos)
{
    int idx = blockIdx.x * blockDim.x + threadIdx.x;
    if (idx >= MROWS * DMODEL) return;
    hs[idx] += pos[idx % (LTOK * DMODEL)];
}

// =====================================================================
// Causal depthwise conv (K=4) + SiLU; rev=1 runs on the reversed sequence.
// Writes f32 (for the scan) and f16 (for the W_x GEMM).
// =====================================================================
__global__ void dwconv_silu_k(const float* __restrict__ xin,
                              const float* __restrict__ w,
                              const float* __restrict__ bias,
                              float* __restrict__ out,
                              _Float16* __restrict__ out16, int rev)
{
    int idx = blockIdx.x * blockDim.x + threadIdx.x;
    if (idx >= MROWS * DMODEL) return;
    int d = idx % DMODEL;
    int t = (idx / DMODEL) % LTOK;
    int b = idx / (DMODEL * LTOK);
    float acc = bias[d];
#pragma unroll
    for (int j = 0; j < KCONV; ++j) {
        int tt = t - (KCONV - 1) + j;
        if (tt >= 0) {
            int l = rev ? (LTOK - 1 - tt) : tt;
            acc += w[d * KCONV + j] * xin[((size_t)(b * LTOK + l)) * DMODEL + d];
        }
    }
    float o = acc / (1.0f + __expf(-acc));   // SiLU
    out[idx]   = o;
    out16[idx] = (_Float16)o;
}

// dt = softplus(dt_pre + b_dt)   (in place)
__global__ void softplus_bias_k(float* __restrict__ dt, const float* __restrict__ bdt)
{
    int idx = blockIdx.x * blockDim.x + threadIdx.x;
    if (idx >= MROWS * DMODEL) return;
    float v = dt[idx] + bdt[idx % DMODEL];
    dt[idx] = (v > 20.f) ? v : log1pf(__expf(v));
}

// =====================================================================
// Selective scan: one lane per (b,d) chain, 16-state SSM in VGPRs.
// =====================================================================
__global__ __launch_bounds__(256) void ssm_scan_k(
    const float* __restrict__ xc, const float* __restrict__ xd,
    const float* __restrict__ dt, const float* __restrict__ Alog,
    const float* __restrict__ Dp, float* __restrict__ y)
{
    int idx = blockIdx.x * blockDim.x + threadIdx.x;
    if (idx >= BDIM * DMODEL) return;
    int d = idx % DMODEL, b = idx / DMODEL;

    float Acoef[NSTATE], h[NSTATE];
#pragma unroll
    for (int n = 0; n < NSTATE; ++n) {
        Acoef[n] = -__expf(Alog[d * NSTATE + n]);
        h[n] = 0.f;
    }
    const float Dd = Dp[d];
    const float* xcp = xc + (size_t)b * LTOK * DMODEL + d;
    const float* dtp = dt + (size_t)b * LTOK * DMODEL + d;
    const float* xdp = xd + (size_t)b * LTOK * XDW;
    float* yp = y + (size_t)b * LTOK * DMODEL + d;

    for (int t = 0; t < LTOK; ++t) {
        float dtv = dtp[(size_t)t * DMODEL];
        float xv  = xcp[(size_t)t * DMODEL];
        const float* row = xdp + (size_t)t * XDW;
        float acc = 0.f;
#pragma unroll
        for (int n = 0; n < NSTATE; ++n) {
            float dA = __expf(dtv * Acoef[n]);                // v_exp_f32 (TRANS)
            h[n] = dA * h[n] + (dtv * row[RRANK + n]) * xv;   // dA*h + dt*B*x
            acc += h[n] * row[RRANK + NSTATE + n];            // h . C
        }
        yp[(size_t)t * DMODEL] = acc + Dd * xv;
    }
}

// u = 0.5*(yf[l] + yb[L-1-l]) * silu(z[l])   -> f16 GEMM operand
__global__ void combine_k(const float* __restrict__ yf, const float* __restrict__ yb,
                          const float* __restrict__ z, _Float16* __restrict__ u)
{
    int idx = blockIdx.x * blockDim.x + threadIdx.x;
    if (idx >= MROWS * DMODEL) return;
    int d = idx % DMODEL;
    int l = (idx / DMODEL) % LTOK;
    int b = idx / (DMODEL * LTOK);
    float zr = z[idx];
    float sg = zr / (1.f + __expf(-zr));
    float vb = yb[((size_t)(b * LTOK + (LTOK - 1 - l))) * DMODEL + d];
    u[idx] = (_Float16)(0.5f * (yf[idx] + vb) * sg);
}

// g = silu(x1) * x2   -> f16 GEMM operand
__global__ void silu_gate_k(const float* __restrict__ x1, const float* __restrict__ x2,
                            _Float16* __restrict__ g, int count)
{
    int idx = blockIdx.x * blockDim.x + threadIdx.x;
    if (idx >= count) return;
    float v = x1[idx];
    g[idx] = (_Float16)((v / (1.f + __expf(-v))) * x2[idx]);
}

// ---------------- host-side helpers ----------------
static inline void launch_gemm(const _Float16* A, int lda, const _Float16* W, int ldw,
                               const float* bias, const float* res,
                               float* Out, int ldo, int M, int Nd, int Kd,
                               hipStream_t s)
{
    int tiles  = ((M + 31) >> 5) * ((Nd + 63) >> 6);
    int blocks = (tiles + 7) / 8;          // 8 waves / 256-thread block
    gemm_wmma_f16<<<blocks, 256, 0, s>>>(A, lda, W, ldw, bias, res, Out, ldo, M, Nd, Kd);
}
static inline int eb(int n) { return (n + 255) / 256; }

extern "C" void kernel_launch(void* const* d_in, const int* in_sizes, int n_in,
                              void* d_out, int out_size, void* d_ws, size_t ws_size,
                              hipStream_t stream)
{
    // ---- inputs (setup_inputs dict order; 'blocks' expands in key order) ----
    const float* x        = (const float*)d_in[0];
    const float* patch_w  = (const float*)d_in[1];   // [D, C*P*P] = [768,768]
    const float* patch_b  = (const float*)d_in[2];
    const float* pos      = (const float*)d_in[3];   // [1,L,D]
    const float* ln1_w    = (const float*)d_in[4];
    const float* ln1_b    = (const float*)d_in[5];
    const float* W_in     = (const float*)d_in[6];   // [DEPTH, 2D, D]
    const float* conv_w   = (const float*)d_in[7];
    const float* conv_b   = (const float*)d_in[8];
    const float* W_x      = (const float*)d_in[9];   // [DEPTH, 80, D]
    const float* W_dt     = (const float*)d_in[10];  // [DEPTH, D, R]
    const float* b_dt     = (const float*)d_in[11];
    const float* A_log    = (const float*)d_in[12];  // [DEPTH, D, N]
    const float* Dp       = (const float*)d_in[13];
    const float* conv_wb  = (const float*)d_in[14];
    const float* conv_bb  = (const float*)d_in[15];
    const float* W_xb     = (const float*)d_in[16];
    const float* W_dtb    = (const float*)d_in[17];
    const float* b_dtb    = (const float*)d_in[18];
    const float* A_logb   = (const float*)d_in[19];
    const float* Dpb      = (const float*)d_in[20];
    const float* W_out    = (const float*)d_in[21];  // [DEPTH, D, D]
    const float* ln2_w    = (const float*)d_in[22];
    const float* ln2_b    = (const float*)d_in[23];
    const float* w1       = (const float*)d_in[24];  // [DEPTH, H, D]
    const float* b1       = (const float*)d_in[25];
    const float* w2       = (const float*)d_in[26];
    const float* b2       = (const float*)d_in[27];
    const float* w3       = (const float*)d_in[28];  // [DEPTH, D, H]
    const float* b3       = (const float*)d_in[29];
    const float* normf_w  = (const float*)d_in[30];
    const float* normf_b  = (const float*)d_in[31];
    float* out = (float*)d_out;

    // ---- workspace carve-up (all offsets multiples of 8 floats -> 16B aligned) ----
    float* ws = (float*)d_ws;
    size_t off = 0;
    const size_t MD = (size_t)MROWS * DMODEL;
    float* hs   = ws + off; off += MD;
    float* xi   = ws + off; off += MD;
    float* zb   = ws + off; off += MD;
    float* xcf  = ws + off; off += MD;
    float* xcb  = ws + off; off += MD;
    float* dtf  = ws + off; off += MD;
    float* dtb  = ws + off; off += MD;
    float* yfb  = ws + off; off += MD;
    float* ybb  = ws + off; off += MD;
    float* xdf  = ws + off; off += (size_t)MROWS * XDW;
    float* xdb  = ws + off; off += (size_t)MROWS * XDW;
    float* x1b  = ws + off; off += (size_t)MROWS * HMLP;
    float* x2b  = ws + off; off += (size_t)MROWS * HMLP;
    _Float16* act16 = (_Float16*)(ws + off); off += (size_t)MROWS * HMLP / 2;   // f16 activation operand
    _Float16* wbuf  = (_Float16*)(ws + off); off += (size_t)HMLP * DMODEL / 2;  // f16 weight operand
    (void)ws_size; (void)n_in; (void)in_sizes; (void)out_size;

    const int NMD = MROWS * DMODEL;

    // ---- patch embed: hs = unfold(x) @ patch_w^T + patch_b + pos ----
    patch_gather_k<<<eb(NMD), 256, 0, stream>>>(x, act16);
    cvt_f16_k<<<eb(DMODEL * DMODEL / 4), 256, 0, stream>>>(patch_w, wbuf, DMODEL * DMODEL / 4);
    launch_gemm(act16, DMODEL, wbuf, DMODEL, patch_b, nullptr,
                hs, DMODEL, MROWS, DMODEL, DMODEL, stream);
    add_pos_k<<<eb(NMD), 256, 0, stream>>>(hs, pos);

    // ---- 12 layers ----
    for (int layer = 0; layer < DEPTH_; ++layer) {
        const float* l1w = ln1_w + (size_t)layer * DMODEL;
        const float* l1b = ln1_b + (size_t)layer * DMODEL;
        const float* Wi  = W_in  + (size_t)layer * 2 * DMODEL * DMODEL;
        const float* cw  = conv_w  + (size_t)layer * DMODEL * KCONV;
        const float* cb  = conv_b  + (size_t)layer * DMODEL;
        const float* cwb = conv_wb + (size_t)layer * DMODEL * KCONV;
        const float* cbb = conv_bb + (size_t)layer * DMODEL;
        const float* Wxl  = W_x  + (size_t)layer * XDW * DMODEL;
        const float* Wxlb = W_xb + (size_t)layer * XDW * DMODEL;
        const float* Wdl  = W_dt  + (size_t)layer * DMODEL * RRANK;
        const float* Wdlb = W_dtb + (size_t)layer * DMODEL * RRANK;
        const float* bdl  = b_dt  + (size_t)layer * DMODEL;
        const float* bdlb = b_dtb + (size_t)layer * DMODEL;
        const float* Al   = A_log  + (size_t)layer * DMODEL * NSTATE;
        const float* Alb  = A_logb + (size_t)layer * DMODEL * NSTATE;
        const float* Dl   = Dp  + (size_t)layer * DMODEL;
        const float* Dlb  = Dpb + (size_t)layer * DMODEL;
        const float* Wo   = W_out + (size_t)layer * DMODEL * DMODEL;
        const float* l2w  = ln2_w + (size_t)layer * DMODEL;
        const float* l2b  = ln2_b + (size_t)layer * DMODEL;
        const float* w1l  = w1 + (size_t)layer * HMLP * DMODEL;
        const float* b1l  = b1 + (size_t)layer * HMLP;
        const float* w2l  = w2 + (size_t)layer * HMLP * DMODEL;
        const float* b2l  = b2 + (size_t)layer * HMLP;
        const float* w3l  = w3 + (size_t)layer * DMODEL * HMLP;
        const float* b3l  = b3 + (size_t)layer * DMODEL;

        // LN1 (-> f16), then W_in split into xi / z (row slices of [2D, D])
        layernorm_k<<<MROWS, 256, 0, stream>>>(hs, l1w, l1b, nullptr, act16);
        cvt_f16_k<<<eb(2 * DMODEL * DMODEL / 4), 256, 0, stream>>>(Wi, wbuf, 2 * DMODEL * DMODEL / 4);
        launch_gemm(act16, DMODEL, wbuf, DMODEL, nullptr, nullptr,
                    xi, DMODEL, MROWS, DMODEL, DMODEL, stream);
        launch_gemm(act16, DMODEL, wbuf + (size_t)DMODEL * DMODEL, DMODEL, nullptr, nullptr,
                    zb, DMODEL, MROWS, DMODEL, DMODEL, stream);

        // forward branch: conv (-> f32 + f16), xd = xc @ W_x^T
        dwconv_silu_k<<<eb(NMD), 256, 0, stream>>>(xi, cw, cb, xcf, act16, 0);
        cvt_f16_k<<<eb(XDW * DMODEL / 4), 256, 0, stream>>>(Wxl, wbuf, XDW * DMODEL / 4);
        launch_gemm(act16, DMODEL, wbuf, DMODEL, nullptr, nullptr,
                    xdf, XDW, MROWS, XDW, DMODEL, stream);

        // backward branch
        dwconv_silu_k<<<eb(NMD), 256, 0, stream>>>(xi, cwb, cbb, xcb, act16, 1);
        cvt_f16_k<<<eb(XDW * DMODEL / 4), 256, 0, stream>>>(Wxlb, wbuf, XDW * DMODEL / 4);
        launch_gemm(act16, DMODEL, wbuf, DMODEL, nullptr, nullptr,
                    xdb, XDW, MROWS, XDW, DMODEL, stream);

        // dt_pre = dtr @ W_dt^T  (K=48 padded to 64, f16 operands)
        pad_f16_k<<<eb(MROWS * RPAD), 256, 0, stream>>>(xdf, XDW, RRANK, act16, RPAD, MROWS * RPAD);
        pad_f16_k<<<eb(DMODEL * RPAD), 256, 0, stream>>>(Wdl, RRANK, RRANK, wbuf, RPAD, DMODEL * RPAD);
        launch_gemm(act16, RPAD, wbuf, RPAD, nullptr, nullptr,
                    dtf, DMODEL, MROWS, DMODEL, RPAD, stream);
        softplus_bias_k<<<eb(NMD), 256, 0, stream>>>(dtf, bdl);

        pad_f16_k<<<eb(MROWS * RPAD), 256, 0, stream>>>(xdb, XDW, RRANK, act16, RPAD, MROWS * RPAD);
        pad_f16_k<<<eb(DMODEL * RPAD), 256, 0, stream>>>(Wdlb, RRANK, RRANK, wbuf, RPAD, DMODEL * RPAD);
        launch_gemm(act16, RPAD, wbuf, RPAD, nullptr, nullptr,
                    dtb, DMODEL, MROWS, DMODEL, RPAD, stream);
        softplus_bias_k<<<eb(NMD), 256, 0, stream>>>(dtb, bdlb);

        // selective scans (fwd and reversed time)
        ssm_scan_k<<<eb(BDIM * DMODEL), 256, 0, stream>>>(xcf, xdf, dtf, Al,  Dl,  yfb);
        ssm_scan_k<<<eb(BDIM * DMODEL), 256, 0, stream>>>(xcb, xdb, dtb, Alb, Dlb, ybb);

        // gate + average both directions (-> f16), then hs += u @ W_out^T
        combine_k<<<eb(NMD), 256, 0, stream>>>(yfb, ybb, zb, act16);
        cvt_f16_k<<<eb(DMODEL * DMODEL / 4), 256, 0, stream>>>(Wo, wbuf, DMODEL * DMODEL / 4);
        launch_gemm(act16, DMODEL, wbuf, DMODEL, nullptr, hs,
                    hs, DMODEL, MROWS, DMODEL, DMODEL, stream);

        // MLP: LN2 (-> f16) -> w1/w2 -> silu-gate (-> f16) -> w3 (+b3, +residual)
        layernorm_k<<<MROWS, 256, 0, stream>>>(hs, l2w, l2b, nullptr, act16);
        cvt_f16_k<<<eb(HMLP * DMODEL / 4), 256, 0, stream>>>(w1l, wbuf, HMLP * DMODEL / 4);
        launch_gemm(act16, DMODEL, wbuf, DMODEL, b1l, nullptr,
                    x1b, HMLP, MROWS, HMLP, DMODEL, stream);
        cvt_f16_k<<<eb(HMLP * DMODEL / 4), 256, 0, stream>>>(w2l, wbuf, HMLP * DMODEL / 4);
        launch_gemm(act16, DMODEL, wbuf, DMODEL, b2l, nullptr,
                    x2b, HMLP, MROWS, HMLP, DMODEL, stream);
        silu_gate_k<<<eb(MROWS * HMLP), 256, 0, stream>>>(x1b, x2b, act16, MROWS * HMLP);
        cvt_f16_k<<<eb(DMODEL * HMLP / 4), 256, 0, stream>>>(w3l, wbuf, DMODEL * HMLP / 4);
        launch_gemm(act16, HMLP, wbuf, HMLP, b3l, hs,
                    hs, DMODEL, MROWS, DMODEL, HMLP, stream);
    }

    // ---- final LN -> output (f32) ----
    layernorm_k<<<MROWS, 256, 0, stream>>>(hs, normf_w, normf_b, out, nullptr);
}
